// Spot_48610439856277
// MI455X (gfx1250) — compile-verified, matrix operand order checked by reading, not run on hardware
//
#include <hip/hip_runtime.h>

typedef __attribute__((ext_vector_type(16))) _Float16 v16h;
typedef __attribute__((ext_vector_type(8)))  float    v8f;

namespace {
constexpr int LEN = 1000;
constexpr int NT  = 256;

// ---- dynamic shared memory layout (in floats) ----
constexpr int OFF_WF   = 0;                       // 1000  waveform
constexpr int OFF_CUM  = OFF_WF   + LEN;          // 1000  cumsum
constexpr int OFF_D1   = OFF_CUM  + LEN;          // 1000  9-tap derivative / out staging
constexpr int OFF_DIV  = OFF_D1   + LEN;          // 1000  51-tap smoothed
constexpr int OFF_W    = OFF_DIV  + LEN;          // 464   w_cum|w_x|w_div|w_roll
constexpr int OFF_WSUB = OFF_W    + 464;          // 945   w_sub [15][3][21]
constexpr int OFF_RED  = OFF_WSUB + 945;          // 256   reduction scratch
constexpr int OFF_SCAN = OFF_RED  + NT;           // 256   scan scratch
constexpr int OFF_COEF = OFF_SCAN + NT;           // 1515  W-folded comb coefs [15][101]
constexpr int OFF_SCAL = OFF_COEF + 1515;         // 16    scalars: W[3], sum_wcum
constexpr int OFF_F1H  = OFF_SCAL + 16;           // 1518  f16 features1_norm [3][1012]
constexpr int OFF_F2   = OFF_F1H  + 1518;         // 17600 features2 [16][1100] (row 15 = dummy)
constexpr int SMEM_FLOATS = OFF_F2 + 16 * 1100;   // ~106.3 KB total
}

// Low 32 bits of a flat LDS-aperture address == LDS byte offset (ISA 10.2).
__device__ __forceinline__ unsigned lds_addr32(const void* p) {
  return (unsigned)(unsigned long long)p;
}

// Async global<->LDS copies (VGLOBAL *_ASYNC_*_LDS_*, tracked by ASYNCcnt).
__device__ __forceinline__ void async_ld_b128(unsigned lds, const void* gaddr) {
  asm volatile("global_load_async_to_lds_b128 %0, %1, off"
               :: "v"(lds), "v"((unsigned long long)gaddr) : "memory");
}
__device__ __forceinline__ void async_ld_b32(unsigned lds, const void* gaddr) {
  asm volatile("global_load_async_to_lds_b32 %0, %1, off"
               :: "v"(lds), "v"((unsigned long long)gaddr) : "memory");
}
__device__ __forceinline__ void async_st_b128(void* gaddr, unsigned lds) {
  asm volatile("global_store_async_from_lds_b128 %0, %1, off"
               :: "v"((unsigned long long)gaddr), "v"(lds) : "memory");
}
__device__ __forceinline__ void wait_async0() {
  asm volatile("s_wait_asynccnt 0" ::: "memory");
}

__device__ __forceinline__ int refl100(int j) {     // reflect for pad<=100, len 1000
  j = (j < 0) ? -j : j;
  return (j >= LEN) ? (2 * LEN - 2 - j) : j;
}

__device__ __forceinline__ float block_reduce_max(float v, float* s_red) {
  const int tid = threadIdx.x;
  s_red[tid] = v; __syncthreads();
  for (int off = NT / 2; off > 0; off >>= 1) {
    if (tid < off) s_red[tid] = fmaxf(s_red[tid], s_red[tid + off]);
    __syncthreads();
  }
  float r = s_red[0]; __syncthreads(); return r;
}

__device__ __forceinline__ float block_reduce_sum(float v, float* s_red) {
  const int tid = threadIdx.x;
  s_red[tid] = v; __syncthreads();
  for (int off = NT / 2; off > 0; off >>= 1) {
    if (tid < off) s_red[tid] += s_red[tid + off];
    __syncthreads();
  }
  float r = s_red[0]; __syncthreads(); return r;
}

// w_sub B-matrix element: q = ci*21 + k (q==63 is zero pad), n = output channel
__device__ __forceinline__ _Float16 wsub_elem(const float* s_wsub, int n, int q) {
  if (n >= 15 || q >= 63) return (_Float16)0.f;   // column 15 zeroed -> dummy D column
  const int ci = q / 21, k = q - ci * 21;
  return (_Float16)s_wsub[n * 63 + ci * 21 + k];
}

// im2col A-matrix element: row m (output position), q = ci*21 + k, reflect pad 10
__device__ __forceinline__ _Float16 f1_elem(const _Float16* s_f1h, int m, int q) {
  if (q >= 63) return (_Float16)0.f;
  const int ci = q / 21, k = q - ci * 21;
  int j = m + k - 10;
  j = (j < 0) ? -j : j;
  if (j >= LEN) j = 2 * LEN - 2 - j;
  return s_f1h[ci * 1012 + j];
}

__global__ __launch_bounds__(NT)
void spot_pipeline(const float* __restrict__ wf, const float* __restrict__ mask,
                   const float* __restrict__ SPE, const float* __restrict__ w_cum,
                   const float* __restrict__ w_x, const float* __restrict__ w_div,
                   const float* __restrict__ w_roll, const float* __restrict__ w_sub,
                   const float* __restrict__ w_comb, const float* __restrict__ W_param,
                   float* __restrict__ out) {
  extern __shared__ float smem[];
  const int tid = threadIdx.x;
  const int b   = blockIdx.x;

  float*     s_wf   = smem + OFF_WF;
  float*     s_cum  = smem + OFF_CUM;
  float*     s_d1   = smem + OFF_D1;
  float*     s_div  = smem + OFF_DIV;
  float*     s_w    = smem + OFF_W;
  float*     s_wsub = smem + OFF_WSUB;
  float*     s_red  = smem + OFF_RED;
  float*     s_scan = smem + OFF_SCAN;
  float*     s_coef = smem + OFF_COEF;
  float*     s_scal = smem + OFF_SCAL;
  _Float16*  s_f1h  = (_Float16*)(smem + OFF_F1H);
  float*     s_f2   = smem + OFF_F2;

  const float* wfrow = wf + (size_t)b * LEN;
  __builtin_prefetch(mask + (size_t)b * LEN + tid, 0, 1);   // global_prefetch_b8

  // ---- stage 0: async DMA waveform + w_sub into LDS; scalar-load small weights ----
  if (tid < 250)                                            // 250 x 16B = 4000B = wf row
    async_ld_b128(lds_addr32(s_wf + tid * 4), wfrow + tid * 4);
  for (int i = tid; i < 945; i += NT)                       // w_sub [15][3][21]
    async_ld_b32(lds_addr32(s_wsub + i), w_sub + i);
  for (int i = tid; i < 121; i += NT) {                     // overlap with async copies
    s_w[i] = w_cum[i]; s_w[121 + i] = w_x[i]; s_w[242 + i] = w_div[i];
  }
  for (int i = tid; i < 101; i += NT) s_w[363 + i] = w_roll[i];
  wait_async0();
  __syncthreads();

  // ---- stage 1: cumsum via two-level block scan (chunks of 4) ----
  {
    float loc = 0.f;
    if (tid < 250) {
      const int base = tid * 4;
      loc = s_wf[base] + s_wf[base + 1] + s_wf[base + 2] + s_wf[base + 3];
    }
    s_scan[tid] = loc;
    __syncthreads();
    for (int off = 1; off < NT; off <<= 1) {
      const float add = (tid >= off) ? s_scan[tid - off] : 0.f;
      __syncthreads();
      s_scan[tid] += add;
      __syncthreads();
    }
    const float prefix = (tid > 0) ? s_scan[tid - 1] : 0.f;
    if (tid < 250) {
      const int base = tid * 4;
      float r = prefix;
      for (int j = 0; j < 4; ++j) { r += s_wf[base + j]; s_cum[base + j] = r; }
    }
    __syncthreads();
  }

  // ---- stage 2: h = max(wf)/max(SPE); W = softmax(-relu(h - W_param)) ----
  float lm = -3.4e38f;
  for (int i = tid; i < LEN; i += NT) lm = fmaxf(lm, s_wf[i]);
  const float wfmax = block_reduce_max(lm, s_red);
  lm = -3.4e38f;
  for (int i = tid; i < LEN; i += NT) lm = fmaxf(lm, SPE[i]);
  const float spemax = block_reduce_max(lm, s_red);
  if (tid == 0) {
    float sw = 0.f;
    for (int k = 0; k < 121; ++k) sw += s_w[k];
    s_scal[3] = sw;                              // sum(w_cum_area)
    const float h = wfmax / spemax;
    float z[3], zm = -3.4e38f;
    for (int c = 0; c < 3; ++c) { z[c] = -fmaxf(h - W_param[c], 0.f); zm = fmaxf(zm, z[c]); }
    float es = 0.f;
    for (int c = 0; c < 3; ++c) { z[c] = expf(z[c] - zm); es += z[c]; }
    for (int c = 0; c < 3; ++c) s_scal[c] = z[c] / es;
  }
  __syncthreads();

  // ---- stage 3: derivative (9-tap, zero pad 4) then running mean (51-tap, zero pad 25) ----
  const float div9[9] = {1.f/280.f, -4.f/105.f, 0.2f, -0.8f, 0.f, 0.8f, -0.2f, 4.f/105.f, -1.f/280.f};
  for (int t = tid; t < LEN; t += NT) {
    float a = 0.f;
    for (int k = 0; k < 9; ++k) {
      const int j = t + k - 4;
      if (j >= 0 && j < LEN) a += div9[k] * s_wf[j];
    }
    s_d1[t] = a;
  }
  __syncthreads();
  for (int t = tid; t < LEN; t += NT) {
    float a = 0.f;
    const int j0 = (t - 25 < 0) ? 0 : t - 25;
    const int j1 = (t + 25 > LEN - 1) ? LEN - 1 : t + 25;
    for (int j = j0; j <= j1; ++j) a += s_d1[j];
    s_div[t] = a * (1.f / 51.f);
  }
  __syncthreads();

  // ---- stage 4: reflect-pad 121/101-tap convs -> features1, channel sums ----
  float ps0 = 0.f, ps1 = 0.f, ps2 = 0.f;
  const float sum_wcum = s_scal[3];
  for (int t = tid; t < LEN; t += NT) {
    float ac = 0.f, ax = 0.f, ad = 0.f, abl = 0.f;
    for (int k = 0; k < 121; ++k) {
      const int j = refl100(t + k - 100);
      ac += s_w[k]       * s_cum[j];
      ax += s_w[121 + k] * s_wf[j];
      ad += s_w[242 + k] * s_div[j];
    }
    for (int k = 0; k < 101; ++k) abl += s_w[363 + k] * s_cum[refl100(t + k - 100)];
    const float fcum = fmaxf(ac - abl * sum_wcum, 0.f);
    const float fx   = fmaxf(ax, 0.f);
    const float fd   = fmaxf(ad, 0.f);
    s_f2[t] = fcum; s_f2[LEN + t] = fx; s_f2[2 * LEN + t] = fd;  // f32 staging in f2 space
    ps0 += fcum; ps1 += fx; ps2 += fd;
  }
  const float S0 = block_reduce_sum(ps0, s_red) + 1e-10f;
  const float S1 = block_reduce_sum(ps1, s_red) + 1e-10f;
  const float S2 = block_reduce_sum(ps2, s_red) + 1e-10f;

  // ---- stage 5: normalize -> f16 features1; fold W into comb coefficients ----
  for (int t = tid; t < LEN; t += NT) {
    s_f1h[t]        = (_Float16)(s_f2[t] / S0);
    s_f1h[1012 + t] = (_Float16)(s_f2[LEN + t] / S1);
    s_f1h[2024 + t] = (_Float16)(s_f2[2 * LEN + t] / S2);
  }
  for (int i = tid; i < 15 * 101; i += NT) {
    const int ch = i / 101, k = i - ch * 101;
    const int c = ch % 3, d = ch / 3;
    s_coef[i] = s_scal[c] * w_comb[d * 303 + k * 3 + c];
  }
  __syncthreads();
  // zero the +/-50 halo of features2 rows 0..14 (for the zero-padded final conv)
  for (int i = tid; i < 15 * 100; i += NT) {
    const int ch = i / 100, r = i - ch * 100;
    s_f2[ch * 1100 + (r < 50 ? r : 1000 + r)] = 0.f;
  }
  __syncthreads();

  // ---- stage 6: 3->15 channel 21-tap conv via WMMA f16 (M=16,N=16,K=64=2x32) ----
  {
    const int lane = tid & 31;
    const int wv   = tid >> 5;      // 8 waves
    const int nn   = lane & 15;     // N (out channel) / M-row index
    const int hi   = lane >> 4;     // lane half selects K sub-block

    v16h b0 = {}, b1 = {};
    #pragma unroll
    for (int e = 0; e < 16; ++e) {
      const int q = ((e < 8) ? e : e + 8) + (hi ? 8 : 0);
      b0[e] = wsub_elem(s_wsub, nn, q);
      b1[e] = wsub_elem(s_wsub, nn, q + 32);
    }
    for (int tile = wv; tile < 63; tile += 8) {       // uniform per wave: EXEC all-1s
      const int m0 = tile * 16;
      const int m  = m0 + nn;
      v16h a0 = {}, a1 = {};
      #pragma unroll
      for (int e = 0; e < 16; ++e) {
        const int q = ((e < 8) ? e : e + 8) + (hi ? 8 : 0);
        a0[e] = f1_elem(s_f1h, m, q);
        a1[e] = f1_elem(s_f1h, m, q + 32);
      }
      v8f c = {};
      c = __builtin_amdgcn_wmma_f32_16x16x32_f16(false, a0, false, b0, (short)0, c, false, false);
      c = __builtin_amdgcn_wmma_f32_16x16x32_f16(false, a1, false, b1, (short)0, c, false, false);

      // D store: no lane guard (row 15 is a dummy); tail guard is a *scalar* branch.
      float* dst = s_f2 + nn * 1100 + 50 + m0 + (hi ? 8 : 0);
      if (__builtin_amdgcn_readfirstlane(tile) != 62) {   // tiles 0..61: contiguous stores
        #pragma unroll
        for (int v = 0; v < 8; ++v) dst[v] = c[v];
      } else {                                            // tile 62: guard pos >= 1000
        #pragma unroll
        for (int v = 0; v < 8; ++v)
          if (m0 + (hi ? 8 : 0) + v < LEN) dst[v] = c[v];
      }
    }
  }
  __syncthreads();

  // ---- stage 7: 15x101 collapse + softplus; then masked, scaled row softmax ----
  float vo[4]; float mk[4];
  int nvals = 0;
  for (int t = tid; t < LEN; t += NT) {
    float acc = 0.f;
    for (int ch = 0; ch < 15; ++ch) {
      const float* f2r = s_f2 + ch * 1100 + t;       // (t+k-50) with +50 halo baked in
      const float* cf  = s_coef + ch * 101;
      for (int k = 0; k < 101; ++k) acc += f2r[k] * cf[k];
    }
    vo[nvals] = fmaxf(acc, 0.f) + log1pf(expf(-fabsf(acc)));  // stable softplus
    mk[nvals] = mask[(size_t)b * LEN + t];
    ++nvals;
  }
  float lmax = -3.4e38f;
  for (int i = 0; i < nvals; ++i) lmax = fmaxf(lmax, vo[i]);
  const float amax = block_reduce_max(lmax, s_red) + 1e-10f;

  float zmax = -3.4e38f;
  for (int i = 0; i < nvals; ++i) {
    vo[i] = 1e4f * (vo[i] / amax) * mk[i];
    zmax = fmaxf(zmax, vo[i]);
  }
  const float zm = block_reduce_max(zmax, s_red);
  float lsum = 0.f;
  for (int i = 0; i < nvals; ++i) { vo[i] = expf(vo[i] - zm); lsum += vo[i]; }
  const float inv = 1.f / block_reduce_sum(lsum, s_red);

  // ---- stage 8: stage result row in LDS, async-store 16B/lane to HBM ----
  int c2 = 0;
  for (int t = tid; t < LEN; t += NT) s_d1[t] = vo[c2++] * inv;
  __syncthreads();
  if (tid < 250)
    async_st_b128(out + (size_t)b * LEN + tid * 4, lds_addr32(s_d1 + tid * 4));
  wait_async0();
}

extern "C" void kernel_launch(void* const* d_in, const int* in_sizes, int n_in,
                              void* d_out, int out_size, void* d_ws, size_t ws_size,
                              hipStream_t stream) {
  (void)n_in; (void)d_ws; (void)ws_size; (void)out_size;
  const float* wf      = (const float*)d_in[0];
  const float* mask    = (const float*)d_in[1];
  const float* SPE     = (const float*)d_in[2];
  const float* w_cum   = (const float*)d_in[3];
  const float* w_x     = (const float*)d_in[4];
  const float* w_div   = (const float*)d_in[5];
  const float* w_roll  = (const float*)d_in[6];
  const float* w_sub   = (const float*)d_in[7];
  const float* w_comb  = (const float*)d_in[8];
  const float* W_param = (const float*)d_in[9];

  const int B = in_sizes[0] / LEN;                  // 4096 samples
  const size_t shmem = (size_t)SMEM_FLOATS * sizeof(float);  // ~106 KB (<320 KB/WG on CDNA5)
  spot_pipeline<<<dim3(B), dim3(NT), shmem, stream>>>(
      wf, mask, SPE, w_cum, w_x, w_div, w_roll, w_sub, w_comb, W_param,
      (float*)d_out);
}
